// GatedCrossScaleFusion_58471684768279
// MI455X (gfx1250) — matrix operand order repaired and checked by reading, not compiled, not run on hardware
//
#include <hip/hip_runtime.h>
#include <hip/hip_bf16.h>
#include <math.h>

typedef __attribute__((ext_vector_type(16))) _Float16 v16h;
typedef __attribute__((ext_vector_type(8)))  float    v8f;
typedef _Float16 half_t;

#define BDIM  4
#define CDIM  256
#define HWDIM 4096
#define RDIM  64
#define EPSV  1e-5f

__device__ __forceinline__ float gelu_exact(float x) {
  return 0.5f * x * (1.0f + erff(x * 0.70710678118654752440f));
}

__device__ __forceinline__ v8f wmma_f16(v16h a, v16h b, v8f c) {
  // D = A(16x32 f16) * B(32x16 f16) + C(16x16 f32)
  return __builtin_amdgcn_wmma_f32_16x16x32_f16(false, a, false, b, (short)0, c,
                                                false, false);
}

// A-matrix 16x32 f16 fragment from a row-major f16 matrix W[ld] (rows = M, cols = K).
// Layout: lanes 0-15: M=lane, v[0..15]=K k0+0..15
//         lanes 16-31: M=lane-16, v[0..15]=K k0+16..31
__device__ __forceinline__ v16h load_afrag_f16(const half_t* __restrict__ W, int ld,
                                               int m0, int k0, int lane) {
  int row = m0 + (lane & 15);
  int kb  = k0 + ((lane >> 4) << 4);
  const half_t* p = W + (size_t)row * ld + kb;
  v16h a;
#pragma unroll
  for (int i = 0; i < 8; ++i) a[i] = p[i];
  p += 8;
#pragma unroll
  for (int i = 0; i < 8; ++i) a[8 + i] = p[i];
  return a;
}

// B-matrix 32x16 f16 fragment from an N-major ("transposed") f16 buffer T:
// element B[k][n] = T[n0+n][k0+k]. Lanes 0-15: col n0+lane, K k0+0..15 (2/VGPR packed);
// lanes 16-31: col n0+lane-16, K k0+16..31 -> contiguous 32B per lane.
__device__ __forceinline__ v16h load_bfrag_f16_T(const half_t* __restrict__ T, int ld,
                                                 int n0, int k0, int lane) {
  int col = n0 + (lane & 15);
  int kb  = k0 + ((lane >> 4) << 4);
  const half_t* p = T + (size_t)col * ld + kb;
  v16h b;
#pragma unroll
  for (int i = 0; i < 16; ++i) b[i] = p[i];
  return b;
}

// B fragment gathered+converted from an fp32 K-major matrix X[K][ldx]: B[k][n]=X[k0+k][n0+n]
__device__ __forceinline__ v16h load_bfrag_f32gather(const float* __restrict__ X, int ldx,
                                                     int k0, int n0, int lane) {
  int col = n0 + (lane & 15);
  int kb  = k0 + ((lane >> 4) << 4);
  v16h b;
#pragma unroll
  for (int i = 0; i < 16; ++i) b[i] = (half_t)X[(size_t)(kb + i) * ldx + col];
  return b;
}

// ---------------------------------------------------------------------------

__global__ void k_f32_to_f16(const float* __restrict__ src, half_t* __restrict__ dst, int n) {
  int i = blockIdx.x * blockDim.x + threadIdx.x;
  if (i < n) dst[i] = (half_t)src[i];
}

__global__ void k_zero(float* __restrict__ p, int n) {
  int i = blockIdx.x * blockDim.x + threadIdx.x;
  if (i < n) p[i] = 0.0f;
}

// D = Wh(f16 [M,Kd]) * X(f32 [Kd,HW]) ; store transposed f16 Out[b][hw][M]
__global__ void k_proj_storeT(const half_t* __restrict__ Wh, const float* __restrict__ X,
                              half_t* __restrict__ Out, int M, int Kd) {
  int lane = threadIdx.x;
  int n0 = blockIdx.x * 16, m0 = blockIdx.y * 16, b = blockIdx.z;
  const float* Xb = X + (size_t)b * Kd * HWDIM;
  half_t* Ob = Out + (size_t)b * HWDIM * M;
  v8f acc = {};
  for (int k0 = 0; k0 < Kd; k0 += 32) {
    v16h a  = load_afrag_f16(Wh, Kd, m0, k0, lane);
    v16h bf = load_bfrag_f32gather(Xb, HWDIM, k0, n0, lane);
    acc = wmma_f16(a, bf, acc);
  }
  // D layout: VGPR j -> M = j + 8*(lane>=16), N = lane&15. Transposed store: 8 contiguous f16.
  int n = n0 + (lane & 15);
  half_t* p = Ob + (size_t)n * M + m0 + ((lane >> 4) << 3);
#pragma unroll
  for (int j = 0; j < 8; ++j) p[j] = (half_t)acc[j];
}

// D = Wh(f16 [M,Kd]) * X(f32 [Kd,HW]) ; store direct f16 Out[b][M][hw]
__global__ void k_proj_direct(const half_t* __restrict__ Wh, const float* __restrict__ X,
                              half_t* __restrict__ Out, int M, int Kd) {
  int lane = threadIdx.x;
  int n0 = blockIdx.x * 16, m0 = blockIdx.y * 16, b = blockIdx.z;
  const float* Xb = X + (size_t)b * Kd * HWDIM;
  half_t* Ob = Out + (size_t)b * M * HWDIM;
  v8f acc = {};
  for (int k0 = 0; k0 < Kd; k0 += 32) {
    v16h a  = load_afrag_f16(Wh, Kd, m0, k0, lane);
    v16h bf = load_bfrag_f32gather(Xb, HWDIM, k0, n0, lane);
    acc = wmma_f16(a, bf, acc);
  }
  int n = n0 + (lane & 15);
  int rbase = (lane >> 4) << 3;
#pragma unroll
  for (int j = 0; j < 8; ++j)
    Ob[(size_t)(m0 + j + rbase) * HWDIM + n] = (half_t)acc[j];
}

// Fused flash attention: softmax(Q Kt) @ V, online softmax, out fp32 [b][c][hw].
// 8 waves / WG; each wave owns 32 output channels; 16-query block per WG.
__global__ void __launch_bounds__(256) k_attention(const half_t* __restrict__ Qt,
                                                   const half_t* __restrict__ Kt,
                                                   const half_t* __restrict__ Vt,
                                                   float* __restrict__ Att) {
  __shared__ _Float16 ldsP[8][16 * 32];  // per-wave P tile (A-layout source)
  int lane = threadIdx.x & 31;
  int w    = threadIdx.x >> 5;
  int n0   = blockIdx.x * 16;
  int b    = blockIdx.y;
  const half_t* Qb = Qt + (size_t)b * HWDIM * RDIM;
  const half_t* Kb = Kt + (size_t)b * HWDIM * RDIM;
  const half_t* Vb = Vt + (size_t)b * CDIM * HWDIM;
  int c0 = w * 32;

  v16h qa0 = load_afrag_f16(Qb, RDIM, n0, 0, lane);   // r = 0..31
  v16h qa1 = load_afrag_f16(Qb, RDIM, n0, 32, lane);  // r = 32..63

  v8f acc0 = {}, acc1 = {};
  float mrow[8], lrow[8];
#pragma unroll
  for (int j = 0; j < 8; ++j) { mrow[j] = -1e30f; lrow[j] = 0.0f; }

  int rbase = (lane >> 4) << 3;
  int cc    = lane & 15;

  for (int m0 = 0; m0 < HWDIM; m0 += 32) {
    // ---- S = Q @ K^T for two 16-col key tiles
    v8f s0 = {}, s1 = {};
    v16h kb0 = load_bfrag_f16_T(Kb, RDIM, m0, 0, lane);
    v16h kb1 = load_bfrag_f16_T(Kb, RDIM, m0, 32, lane);
    s0 = wmma_f16(qa0, kb0, s0);
    s0 = wmma_f16(qa1, kb1, s0);
    v16h kb2 = load_bfrag_f16_T(Kb, RDIM, m0 + 16, 0, lane);
    v16h kb3 = load_bfrag_f16_T(Kb, RDIM, m0 + 16, 32, lane);
    s1 = wmma_f16(qa0, kb2, s1);
    s1 = wmma_f16(qa1, kb3, s1);

    if (m0 + 32 < HWDIM)
      __builtin_prefetch(Kb + (size_t)(m0 + 32) * RDIM, 0, 0);  // global_prefetch_b8

    // ---- online softmax (row reductions stay in 16-lane halves: xor 1/2/4/8)
#pragma unroll
    for (int j = 0; j < 8; ++j) {
      float v = fmaxf(s0[j], s1[j]);
#pragma unroll
      for (int off = 1; off < 16; off <<= 1) v = fmaxf(v, __shfl_xor(v, off, 32));
      float mnew = fmaxf(mrow[j], v);
      float corr = __expf(mrow[j] - mnew);
      float p0 = __expf(s0[j] - mnew);
      float p1 = __expf(s1[j] - mnew);
      float rs = p0 + p1;
#pragma unroll
      for (int off = 1; off < 16; off <<= 1) rs += __shfl_xor(rs, off, 32);
      lrow[j] = lrow[j] * corr + rs;
      mrow[j] = mnew;
      acc0[j] *= corr;
      acc1[j] *= corr;
      // D-layout -> row-major P[16][32] in this wave's LDS slice (f16)
      ldsP[w][(j + rbase) * 32 + cc]      = (half_t)p0;
      ldsP[w][(j + rbase) * 32 + 16 + cc] = (half_t)p1;
    }

    // ---- O += P @ V for this wave's two channel tiles
    v16h pa  = load_afrag_f16(&ldsP[w][0], 32, 0, 0, lane);
    v16h vb0 = load_bfrag_f16_T(Vb, HWDIM, c0, m0, lane);
    v16h vb1 = load_bfrag_f16_T(Vb, HWDIM, c0 + 16, m0, lane);
    acc0 = wmma_f16(pa, vb0, acc0);
    acc1 = wmma_f16(pa, vb1, acc1);
  }

  int n = n0 + (lane & 15);
#pragma unroll
  for (int j = 0; j < 8; ++j) {
    float inv = 1.0f / lrow[j];
    Att[((size_t)b * CDIM + c0 + j + rbase) * HWDIM + n]      = acc0[j] * inv;
    Att[((size_t)b * CDIM + c0 + 16 + j + rbase) * HWDIM + n] = acc1[j] * inv;
  }
}

// g = gelu(Wg1 @ concat(dec, att)); accumulate per-(b,c) spatial sums for the mean pool.
__global__ void k_gate1(const half_t* __restrict__ Wg1h, const float* __restrict__ Dec,
                        const float* __restrict__ Att, float* __restrict__ gsum) {
  int lane = threadIdx.x;
  int n0 = blockIdx.x * 16, c0 = blockIdx.y * 16, b = blockIdx.z;
  const float* Db = Dec + (size_t)b * CDIM * HWDIM;
  const float* Ab = Att + (size_t)b * CDIM * HWDIM;
  int col = n0 + (lane & 15);
  int kb  = (lane >> 4) << 4;
  v8f acc = {};
  for (int k0 = 0; k0 < 2 * CDIM; k0 += 32) {
    v16h a = load_afrag_f16(Wg1h, 2 * CDIM, c0, k0, lane);
    v16h bf;
#pragma unroll
    for (int i = 0; i < 16; ++i) {
      int kk = k0 + kb + i;
      float xv = (kk < CDIM) ? Db[(size_t)kk * HWDIM + col]
                             : Ab[(size_t)(kk - CDIM) * HWDIM + col];
      bf[i] = (half_t)xv;
    }
    acc = wmma_f16(a, bf, acc);
  }
  int rbase = (lane >> 4) << 3;
#pragma unroll
  for (int j = 0; j < 8; ++j) {
    float g = gelu_exact(acc[j]);
#pragma unroll
    for (int off = 1; off < 16; off <<= 1) g += __shfl_xor(g, off, 32);
    if ((lane & 15) == 0) atomicAdd(&gsum[b * CDIM + c0 + j + rbase], g);
  }
}

// gate = sigmoid(Wg2 @ (gsum/HW) + bg2)
__global__ void k_gate2(const float* __restrict__ Wg2, const float* __restrict__ bg2,
                        const float* __restrict__ gsum, float* __restrict__ gate) {
  int b = blockIdx.x;
  int c = threadIdx.x;
  const float invHW = 1.0f / (float)HWDIM;
  float acc = bg2[c];
  for (int k = 0; k < CDIM; ++k)
    acc += Wg2[c * CDIM + k] * (gsum[b * CDIM + k] * invHW);
  gate[b * CDIM + c] = 1.0f / (1.0f + __expf(-acc));
}

// y = Wo @ (gate*att + (1-gate)*dec); write y fp32 and accumulate BN stats.
__global__ void k_wo(const half_t* __restrict__ Woh, const float* __restrict__ Dec,
                     const float* __restrict__ Att, const float* __restrict__ gate,
                     float* __restrict__ Y, float* __restrict__ ysum,
                     float* __restrict__ ysumsq) {
  int lane = threadIdx.x;
  int n0 = blockIdx.x * 16, c0 = blockIdx.y * 16, b = blockIdx.z;
  const float* Db = Dec + (size_t)b * CDIM * HWDIM;
  const float* Ab = Att + (size_t)b * CDIM * HWDIM;
  const float* gb = gate + b * CDIM;
  int col = n0 + (lane & 15);
  int kb  = (lane >> 4) << 4;
  v8f acc = {};
  for (int k0 = 0; k0 < CDIM; k0 += 32) {
    v16h a = load_afrag_f16(Woh, CDIM, c0, k0, lane);
    v16h bf;
#pragma unroll
    for (int i = 0; i < 16; ++i) {
      int kk = k0 + kb + i;
      float g = gb[kk];
      float xv = g * Ab[(size_t)kk * HWDIM + col] + (1.0f - g) * Db[(size_t)kk * HWDIM + col];
      bf[i] = (half_t)xv;
    }
    acc = wmma_f16(a, bf, acc);
  }
  int rbase = (lane >> 4) << 3;
#pragma unroll
  for (int j = 0; j < 8; ++j) {
    float yv = acc[j];
    Y[((size_t)b * CDIM + c0 + j + rbase) * HWDIM + col] = yv;
    float s = yv, sq = yv * yv;
#pragma unroll
    for (int off = 1; off < 16; off <<= 1) {
      s  += __shfl_xor(s, off, 32);
      sq += __shfl_xor(sq, off, 32);
    }
    if ((lane & 15) == 0) {
      atomicAdd(&ysum[c0 + j + rbase], s);
      atomicAdd(&ysumsq[c0 + j + rbase], sq);
    }
  }
}

__global__ void k_bn_gelu(const float* __restrict__ Y, const float* __restrict__ ysum,
                          const float* __restrict__ ysumsq, const float* __restrict__ gamma,
                          const float* __restrict__ beta, float* __restrict__ out) {
  int idx = blockIdx.x * blockDim.x + threadIdx.x;
  if (idx >= BDIM * CDIM * HWDIM) return;
  int c = (idx / HWDIM) % CDIM;
  const float invN = 1.0f / (float)(BDIM * HWDIM);
  float mean = ysum[c] * invN;
  float var  = ysumsq[c] * invN - mean * mean;
  float yn   = (Y[idx] - mean) * rsqrtf(var + EPSV);
  out[idx] = gelu_exact(yn * gamma[c] + beta[c]);
}

__global__ void k_attn_map(const float* __restrict__ Att, float* __restrict__ out) {
  int idx = blockIdx.x * blockDim.x + threadIdx.x;  // b*HW + n
  if (idx >= BDIM * HWDIM) return;
  int b = idx / HWDIM, n = idx % HWDIM;
  float s = 0.0f;
  for (int c = 0; c < CDIM; ++c) s += Att[((size_t)b * CDIM + c) * HWDIM + n];
  out[idx] = s * (1.0f / (float)CDIM);
}

// ---------------------------------------------------------------------------

extern "C" void kernel_launch(void* const* d_in, const int* in_sizes, int n_in,
                              void* d_out, int out_size, void* d_ws, size_t ws_size,
                              hipStream_t stream) {
  (void)in_sizes; (void)n_in; (void)out_size; (void)ws_size;
  const float* dec   = (const float*)d_in[0];
  const float* enc   = (const float*)d_in[1];
  const float* Wq    = (const float*)d_in[2];
  const float* Wk    = (const float*)d_in[3];
  const float* Wv    = (const float*)d_in[4];
  const float* Wg1   = (const float*)d_in[5];
  const float* Wg2   = (const float*)d_in[6];
  const float* bg2   = (const float*)d_in[7];
  const float* Wo    = (const float*)d_in[8];
  const float* gamma = (const float*)d_in[9];
  const float* beta  = (const float*)d_in[10];
  float* out = (float*)d_out;

  char* ws = (char*)d_ws;
  size_t off = 0;
  auto walloc = [&](size_t bytes) -> void* {
    void* p = ws + off;
    off += (bytes + 255) & ~(size_t)255;
    return p;
  };
  half_t* Qt     = (half_t*)walloc((size_t)BDIM * HWDIM * RDIM * 2);   // [b,hw,64]
  half_t* Ktb    = (half_t*)walloc((size_t)BDIM * HWDIM * RDIM * 2);   // [b,hw,64]
  half_t* Vt     = (half_t*)walloc((size_t)BDIM * CDIM * HWDIM * 2);   // [b,c,hw]
  float*  AttW   = (float*) walloc((size_t)BDIM * CDIM * HWDIM * 4);   // attended [b,c,hw]
  float*  Yw     = (float*) walloc((size_t)BDIM * CDIM * HWDIM * 4);   // pre-BN y
  half_t* Wqh    = (half_t*)walloc((size_t)RDIM * CDIM * 2);
  half_t* Wkh    = (half_t*)walloc((size_t)RDIM * CDIM * 2);
  half_t* Wvh    = (half_t*)walloc((size_t)CDIM * CDIM * 2);
  half_t* Wg1h   = (half_t*)walloc((size_t)CDIM * 2 * CDIM * 2);
  half_t* Woh    = (half_t*)walloc((size_t)CDIM * CDIM * 2);
  float*  gsum   = (float*) walloc((size_t)BDIM * CDIM * 4);
  float*  gateb  = (float*) walloc((size_t)BDIM * CDIM * 4);
  float*  ysum   = (float*) walloc((size_t)CDIM * 4);
  float*  ysumsq = (float*) walloc((size_t)CDIM * 4);

  auto cvt = [&](const float* s, half_t* d, int n) {
    k_f32_to_f16<<<dim3((n + 255) / 256), dim3(256), 0, stream>>>(s, d, n);
  };
  cvt(Wq, Wqh, RDIM * CDIM);
  cvt(Wk, Wkh, RDIM * CDIM);
  cvt(Wv, Wvh, CDIM * CDIM);
  cvt(Wg1, Wg1h, CDIM * 2 * CDIM);
  cvt(Wo, Woh, CDIM * CDIM);

  k_zero<<<dim3((BDIM * CDIM + 255) / 256), dim3(256), 0, stream>>>(gsum, BDIM * CDIM);
  k_zero<<<dim3(1), dim3(256), 0, stream>>>(ysum, CDIM);
  k_zero<<<dim3(1), dim3(256), 0, stream>>>(ysumsq, CDIM);

  // Projections (WMMA): Q from decoder, K/V from encoder
  k_proj_storeT<<<dim3(HWDIM / 16, RDIM / 16, BDIM), dim3(32), 0, stream>>>(
      Wqh, dec, Qt, RDIM, CDIM);
  k_proj_storeT<<<dim3(HWDIM / 16, RDIM / 16, BDIM), dim3(32), 0, stream>>>(
      Wkh, enc, Ktb, RDIM, CDIM);
  k_proj_direct<<<dim3(HWDIM / 16, CDIM / 16, BDIM), dim3(32), 0, stream>>>(
      Wvh, enc, Vt, CDIM, CDIM);

  // Fused flash attention (WMMA QK^T + online softmax + WMMA PV)
  k_attention<<<dim3(HWDIM / 16, BDIM), dim3(256), 0, stream>>>(Qt, Ktb, Vt, AttW);

  // Gate MLP
  k_gate1<<<dim3(HWDIM / 16, CDIM / 16, BDIM), dim3(32), 0, stream>>>(Wg1h, dec, AttW, gsum);
  k_gate2<<<dim3(BDIM), dim3(CDIM), 0, stream>>>(Wg2, bg2, gsum, gateb);

  // Output conv + BN stats
  k_wo<<<dim3(HWDIM / 16, CDIM / 16, BDIM), dim3(32), 0, stream>>>(
      Woh, dec, AttW, gateb, Yw, ysum, ysumsq);

  // BN + GELU epilogue -> out[0], attn map -> out[1]
  k_bn_gelu<<<dim3((BDIM * CDIM * HWDIM) / 256), dim3(256), 0, stream>>>(
      Yw, ysum, ysumsq, gamma, beta, out);
  k_attn_map<<<dim3((BDIM * HWDIM) / 256), dim3(256), 0, stream>>>(
      AttW, out + (size_t)BDIM * CDIM * HWDIM);
}
